// SpatioTemporalRNNDiTBlock_34660386079344
// MI455X (gfx1250) — compile-verified, matrix operand order checked by reading, not run on hardware
//
#include <hip/hip_runtime.h>
#include <hip/hip_bf16.h>
#include <math.h>
#include <stdint.h>

// ---------------------------------------------------------------------------
// Types for CDNA5 WMMA (gfx1250, wave32)
// ---------------------------------------------------------------------------
typedef __bf16 bf16_t;
typedef __attribute__((ext_vector_type(16))) __bf16 v16bf;
typedef __attribute__((ext_vector_type(8)))  __bf16 v8bf;
typedef __attribute__((ext_vector_type(4)))  __bf16 v4bf;
typedef __attribute__((ext_vector_type(8)))  float  v8f;

#define WMMA_BF16(a, b, c)                                                     \
    __builtin_amdgcn_wmma_f32_16x16x32_bf16(false, (a), false, (b), (short)0,  \
                                            (c), false, false)

// Load a 16x32 bf16 fragment from an LDS tile stored row-major with stride 32.
// CDNA5 16-bit A-matrix layout: lanes 0-15 hold row M=lane with K={0..7,16..23};
// lanes 16-31 hold row M=lane-16 with K={8..15,24..31}. The B fragment uses the
// identical pattern when the B tile is stored N-major (transposed) in LDS.
static __device__ __forceinline__ v16bf frag_ld(const bf16_t* tile)
{
    const int lane = threadIdx.x & 31;
    const bf16_t* p = tile + (lane & 15) * 32 + ((lane >> 4) << 3);
    v8bf lo = *(const v8bf*)(p);
    v8bf hi = *(const v8bf*)(p + 16);
    v16bf f;
#pragma unroll
    for (int i = 0; i < 8; ++i) { f[i] = lo[i]; f[i + 8] = hi[i]; }
    return f;
}

// Async DMA of 16 bytes per lane, global -> LDS (CDNA5 GLOBAL_LOAD_ASYNC_TO_LDS,
// tracked by ASYNCcnt; ISA 15.18.3 op 98). ldsoff = LDS byte offset (low 32
// bits of the generic shared-space address per the flat-aperture mapping).
static __device__ __forceinline__ void async_cp_b128(unsigned ldsoff, const bf16_t* gaddr)
{
    asm volatile("global_load_async_to_lds_b128 %0, %1, off"
                 :: "v"(ldsoff), "v"(gaddr) : "memory");
}

static __device__ __forceinline__ void async_wait0()
{
    asm volatile("s_wait_asynccnt 0x0" ::: "memory");
}

static __device__ __forceinline__ float gelu_tanh(float x)
{
    float x3 = x * x * x;
    return 0.5f * x * (1.0f + tanhf(0.7978845608028654f * (x + 0.044715f * x3)));
}

static __device__ __forceinline__ float sigm(float x)
{
    return 1.0f / (1.0f + __expf(-x));
}

// ---------------------------------------------------------------------------
// Weight convert+transpose: W[K,N] f32  ->  WT[N,K] bf16.  K,N multiples of 64.
// 64x64 tile through LDS; coalesced b128 reads and b64 writes.
// ---------------------------------------------------------------------------
__global__ __launch_bounds__(256) void wconv_kernel(
    const float* __restrict__ W, bf16_t* __restrict__ WT, int K, int N)
{
    __shared__ float tile[64][68];
    const int k0 = blockIdx.y * 64, n0 = blockIdx.x * 64;
    const int tid = threadIdx.x;
    const int tr  = tid >> 4;          // 0..15
    const int tc4 = (tid & 15) * 4;    // 0..60
#pragma unroll
    for (int j = 0; j < 4; ++j) {
        int kr = tr + j * 16;
        float4 v = *(const float4*)(W + (size_t)(k0 + kr) * N + n0 + tc4);
        tile[kr][tc4 + 0] = v.x; tile[kr][tc4 + 1] = v.y;
        tile[kr][tc4 + 2] = v.z; tile[kr][tc4 + 3] = v.w;
    }
    __syncthreads();
#pragma unroll
    for (int j = 0; j < 4; ++j) {
        int nr = tr + j * 16;
        v4bf b = { (bf16_t)tile[tc4 + 0][nr], (bf16_t)tile[tc4 + 1][nr],
                   (bf16_t)tile[tc4 + 2][nr], (bf16_t)tile[tc4 + 3][nr] };
        *(v4bf*)(WT + (size_t)(n0 + nr) * K + k0 + tc4) = b;
    }
}

// ---------------------------------------------------------------------------
// WMMA GEMM:  C[M,N] = act(A[M,K] @ W[K,N] + bias0 + bias1)
// A: bf16 [M,K] row-major.  BT: bf16 [N,K] row-major (pre-transposed weight).
// 256 threads = 8 waves; block tile 128x128, K-step 32; each wave computes a
// 32x64 tile with 8 v_wmma per K-step. Interior blocks stage with the async
// global->LDS DMA engine (16B per lane, ASYNCcnt-tracked), no VGPR round-trip.
// OB=0: f32 output in Cf; OB=1: bf16 output in Cb.
// ---------------------------------------------------------------------------
template <int ACT, int OB>
__global__ __launch_bounds__(256) void gemm_kernel(
    const bf16_t* __restrict__ A, int lda,
    const bf16_t* __restrict__ BT, int ldb,
    const float* __restrict__ bias0, const float* __restrict__ bias1,
    float* __restrict__ Cf, bf16_t* __restrict__ Cb,
    int ldc, int M, int N, int K)
{
    __shared__ bf16_t As[128 * 32];
    __shared__ bf16_t Bs[128 * 32];   // Bs[n][k]

    const int tid = threadIdx.x;
    const int n0  = blockIdx.x * 128;
    const int m0  = blockIdx.y * 128;
    const int wid = tid >> 5;
    const int wm  = (wid >> 1) * 32;   // 0,32,64,96
    const int wn  = (wid & 1) * 64;    // 0,64
    const bool full = ((K & 31) == 0) && (m0 + 128 <= M) && (n0 + 128 <= N);

    const unsigned ldsA = (unsigned)(uintptr_t)(void*)As;
    const unsigned ldsB = (unsigned)(uintptr_t)(void*)Bs;

    v8f acc[2][4] = {};

    for (int k0 = 0; k0 < K; k0 += 32) {
        if (full) {
            // A tile 128x32 bf16 = 512 16-byte chunks, 2 per thread (async DMA)
#pragma unroll
            for (int j = 0; j < 2; ++j) {
                int cidx = tid + j * 256;
                int r = cidx >> 2, c8 = (cidx & 3) << 3;
                async_cp_b128(ldsA + (unsigned)(r * 32 + c8) * 2,
                              A + (size_t)(m0 + r) * lda + k0 + c8);
            }
            // B tile: identical contiguous copy from pre-transposed weight
#pragma unroll
            for (int j = 0; j < 2; ++j) {
                int cidx = tid + j * 256;
                int r = cidx >> 2, c8 = (cidx & 3) << 3;
                async_cp_b128(ldsB + (unsigned)(r * 32 + c8) * 2,
                              BT + (size_t)(n0 + r) * ldb + k0 + c8);
            }
            if (k0 + 32 < K) {
                __builtin_prefetch(A + (size_t)(m0 + (tid >> 1)) * lda + k0 + 32, 0, 1);
                __builtin_prefetch(BT + (size_t)(n0 + (tid >> 1)) * ldb + k0 + 32, 0, 1);
            }
            async_wait0();   // this wave's DMAs landed; barrier covers the rest
        } else {
            for (int i = tid; i < 128 * 32; i += 256) {
                int r = i >> 5, cc = i & 31;
                bf16_t v = (bf16_t)0.0f;
                if (m0 + r < M && k0 + cc < K) v = A[(size_t)(m0 + r) * lda + k0 + cc];
                As[i] = v;
            }
            for (int i = tid; i < 128 * 32; i += 256) {
                int r = i >> 5, cc = i & 31;
                bf16_t v = (bf16_t)0.0f;
                if (n0 + r < N && k0 + cc < K) v = BT[(size_t)(n0 + r) * ldb + k0 + cc];
                Bs[i] = v;
            }
        }
        __syncthreads();

        v16bf a0 = frag_ld(As + wm * 32);
        v16bf a1 = frag_ld(As + (wm + 16) * 32);
#pragma unroll
        for (int j = 0; j < 4; ++j) {
            v16bf bj = frag_ld(Bs + (wn + j * 16) * 32);
            acc[0][j] = WMMA_BF16(a0, bj, acc[0][j]);
            acc[1][j] = WMMA_BF16(a1, bj, acc[1][j]);
        }
        __syncthreads();
    }

    // epilogue (C/D layout: lane l, vgpr r -> row = r + 8*(l>>4), col = l&15)
    const int lane = tid & 31;
    const int rsub = (lane >> 4) * 8;
    const int csub = lane & 15;
#pragma unroll
    for (int i = 0; i < 2; ++i)
#pragma unroll
        for (int j = 0; j < 4; ++j)
#pragma unroll
            for (int rr = 0; rr < 8; ++rr) {
                int row = m0 + wm + i * 16 + rr + rsub;
                int col = n0 + wn + j * 16 + csub;
                if (row < M && col < N) {
                    float v = acc[i][j][rr];
                    if (bias0) v += bias0[col];
                    if (bias1) v += bias1[col];
                    if (ACT == 1) v = gelu_tanh(v);
                    if (OB) Cb[(size_t)row * ldc + col] = (bf16_t)v;
                    else    Cf[(size_t)row * ldc + col] = v;
                }
            }
}

// ---------------------------------------------------------------------------
// Attention scores: S[z,q,k] = (Q_h q).(K_h k)/8 (f32 out), optional causal.
// qkv (bf16) layout: [inst, L, 3*1024]; head h: Q at h*64, K at 1024+h*64,
// V at 2048+h*64. z = inst*16 + h. 128 threads, 64x64 tile, dh=64.
// ---------------------------------------------------------------------------
template <bool CAUSAL>
__global__ __launch_bounds__(128) void attn_score_kernel(
    const bf16_t* __restrict__ qkv, int L, float* __restrict__ S)
{
    const int z    = blockIdx.z;
    const int inst = z >> 4;
    const int h    = z & 15;
    const bf16_t* Q  = qkv + (size_t)inst * L * 3072 + h * 64;
    const bf16_t* Km = Q + 1024;
    const int m0 = blockIdx.y * 64;   // query tile
    const int n0 = blockIdx.x * 64;   // key tile

    __shared__ bf16_t As[64 * 32];
    __shared__ bf16_t Bs[64 * 32];

    const int tid = threadIdx.x;
    const int wid = tid >> 5;
    const int wm  = (wid >> 1) * 32;
    const int wn  = (wid & 1) * 32;
    const bool fullA = (m0 + 64 <= L);
    const bool fullB = (n0 + 64 <= L);

    v8f acc[2][2] = {};

    for (int k0 = 0; k0 < 64; k0 += 32) {
        if (fullA) {
#pragma unroll
            for (int j = 0; j < 2; ++j) {
                int cidx = tid + j * 128;   // 256 chunks of 8 (64x32)
                int r = cidx >> 2, c8 = (cidx & 3) << 3;
                v8bf v = *(const v8bf*)(Q + (size_t)(m0 + r) * 3072 + k0 + c8);
                *(v8bf*)(As + r * 32 + c8) = v;
            }
        } else {
            for (int i = tid; i < 64 * 32; i += 128) {
                int r = i >> 5, cc = i & 31;
                As[i] = (m0 + r < L) ? Q[(size_t)(m0 + r) * 3072 + k0 + cc] : (bf16_t)0.0f;
            }
        }
        if (fullB) {
#pragma unroll
            for (int j = 0; j < 2; ++j) {
                int cidx = tid + j * 128;
                int r = cidx >> 2, c8 = (cidx & 3) << 3;
                v8bf v = *(const v8bf*)(Km + (size_t)(n0 + r) * 3072 + k0 + c8);
                *(v8bf*)(Bs + r * 32 + c8) = v;
            }
        } else {
            for (int i = tid; i < 64 * 32; i += 128) {
                int r = i >> 5, cc = i & 31;
                Bs[i] = (n0 + r < L) ? Km[(size_t)(n0 + r) * 3072 + k0 + cc] : (bf16_t)0.0f;
            }
        }
        __syncthreads();
        v16bf a0 = frag_ld(As + wm * 32);
        v16bf a1 = frag_ld(As + (wm + 16) * 32);
        v16bf b0 = frag_ld(Bs + wn * 32);
        v16bf b1 = frag_ld(Bs + (wn + 16) * 32);
        acc[0][0] = WMMA_BF16(a0, b0, acc[0][0]);
        acc[0][1] = WMMA_BF16(a0, b1, acc[0][1]);
        acc[1][0] = WMMA_BF16(a1, b0, acc[1][0]);
        acc[1][1] = WMMA_BF16(a1, b1, acc[1][1]);
        __syncthreads();
    }

    const int lane = tid & 31;
    const int rsub = (lane >> 4) * 8;
    const int csub = lane & 15;
#pragma unroll
    for (int i = 0; i < 2; ++i)
#pragma unroll
        for (int j = 0; j < 2; ++j)
#pragma unroll
            for (int rr = 0; rr < 8; ++rr) {
                int row = m0 + wm + i * 16 + rr + rsub;
                int col = n0 + wn + j * 16 + csub;
                if (row < L && col < L) {
                    float v = acc[i][j][rr] * 0.125f;   // 1/sqrt(64)
                    if (CAUSAL && col > row) v = -3.402823466e38f;
                    S[(size_t)z * L * L + (size_t)row * L + col] = v;
                }
            }
}

// ---------------------------------------------------------------------------
// P @ V:  O[inst, q, h*64+d] = sum_k P[z,q,k] * V[inst,k,h*64+d]; bf16 out.
// ---------------------------------------------------------------------------
__global__ __launch_bounds__(128) void attn_av_kernel(
    const bf16_t* __restrict__ P, const bf16_t* __restrict__ qkv, int L,
    bf16_t* __restrict__ O)
{
    const int z    = blockIdx.z;
    const int inst = z >> 4;
    const int h    = z & 15;
    const bf16_t* Pz = P + (size_t)z * L * L;
    const bf16_t* V  = qkv + (size_t)inst * L * 3072 + 2048 + h * 64;
    const int m0 = blockIdx.y * 64;   // query rows; N tile is the full dh=64

    __shared__ bf16_t As[64 * 32];
    __shared__ bf16_t Bs[64 * 32];

    const int tid = threadIdx.x;
    const int wid = tid >> 5;
    const int wm  = (wid >> 1) * 32;
    const int wn  = (wid & 1) * 32;

    v8f acc[2][2] = {};

    for (int k0 = 0; k0 < L; k0 += 32) {
        const bool fullA = (m0 + 64 <= L) && (k0 + 32 <= L);
        const bool fullV = (k0 + 32 <= L);
        if (fullA) {
#pragma unroll
            for (int j = 0; j < 2; ++j) {
                int cidx = tid + j * 128;
                int r = cidx >> 2, c8 = (cidx & 3) << 3;
                v8bf v = *(const v8bf*)(Pz + (size_t)(m0 + r) * L + k0 + c8);
                *(v8bf*)(As + r * 32 + c8) = v;
            }
        } else {
            for (int i = tid; i < 64 * 32; i += 128) {
                int r = i >> 5, cc = i & 31;
                bf16_t v = (bf16_t)0.0f;
                if (m0 + r < L && k0 + cc < L) v = Pz[(size_t)(m0 + r) * L + k0 + cc];
                As[i] = v;
            }
        }
        if (fullV) {
            // V tile 32(k) x 64(d): v8bf loads along d, transposed b16 stores
#pragma unroll
            for (int j = 0; j < 2; ++j) {
                int cidx = tid + j * 128;            // 256 chunks
                int kk = cidx >> 3, d8 = (cidx & 7) << 3;
                v8bf v = *(const v8bf*)(V + (size_t)(k0 + kk) * 3072 + d8);
#pragma unroll
                for (int e = 0; e < 8; ++e) Bs[(d8 + e) * 32 + kk] = v[e];
            }
        } else {
            for (int i = tid; i < 32 * 64; i += 128) {
                int kk = i >> 6, nn = i & 63;
                bf16_t v = (bf16_t)0.0f;
                if (k0 + kk < L) v = V[(size_t)(k0 + kk) * 3072 + nn];
                Bs[nn * 32 + kk] = v;
            }
        }
        __syncthreads();
        v16bf a0 = frag_ld(As + wm * 32);
        v16bf a1 = frag_ld(As + (wm + 16) * 32);
        v16bf b0 = frag_ld(Bs + wn * 32);
        v16bf b1 = frag_ld(Bs + (wn + 16) * 32);
        acc[0][0] = WMMA_BF16(a0, b0, acc[0][0]);
        acc[0][1] = WMMA_BF16(a0, b1, acc[0][1]);
        acc[1][0] = WMMA_BF16(a1, b0, acc[1][0]);
        acc[1][1] = WMMA_BF16(a1, b1, acc[1][1]);
        __syncthreads();
    }

    const int lane = tid & 31;
    const int rsub = (lane >> 4) * 8;
    const int csub = lane & 15;
#pragma unroll
    for (int i = 0; i < 2; ++i)
#pragma unroll
        for (int j = 0; j < 2; ++j)
#pragma unroll
            for (int rr = 0; rr < 8; ++rr) {
                int row = m0 + wm + i * 16 + rr + rsub;
                int col = wn + j * 16 + csub;   // 0..63 == dh
                if (row < L)
                    O[((size_t)inst * L + row) * 1024 + h * 64 + col] = (bf16_t)acc[i][j][rr];
            }
}

// ---------------------------------------------------------------------------
// Elementwise / reduction kernels
// ---------------------------------------------------------------------------
__global__ void silu_kernel(const float4* __restrict__ c, bf16_t* __restrict__ o, int n4)
{
    int i = blockIdx.x * blockDim.x + threadIdx.x;
    if (i < n4) {
        float4 v = c[i];
        v4bf b = { (bf16_t)(v.x * sigm(v.x)), (bf16_t)(v.y * sigm(v.y)),
                   (bf16_t)(v.z * sigm(v.z)), (bf16_t)(v.w * sigm(v.w)) };
        *(v4bf*)(o + (size_t)i * 4) = b;
    }
}

__global__ void copy_kernel(const float4* __restrict__ a, float4* __restrict__ b, int n4)
{
    int i = blockIdx.x * blockDim.x + threadIdx.x;
    if (i < n4) b[i] = a[i];
}

__global__ void zero_kernel(float4* __restrict__ a, int n4)
{
    int i = blockIdx.x * blockDim.x + threadIdx.x;
    if (i < n4) a[i] = make_float4(0.f, 0.f, 0.f, 0.f);
}

// LayerNorm(1024) + modulate, bf16 output; one block (256 thr) per token row.
// transposed: write row (b*256 + s)*16 + t instead of r = b*4096 + t*256 + s.
__global__ __launch_bounds__(256) void ln_mod_kernel(
    const float* __restrict__ x, const float* __restrict__ modbuf, int mod_ld,
    int shift_off, int scale_off, bf16_t* __restrict__ out, int transposed)
{
    const int r   = blockIdx.x;
    const int tid = threadIdx.x;
    const float* xr = x + (size_t)r * 1024;
    __shared__ float red[256];

    float4 v = *(const float4*)(xr + tid * 4);

    red[tid] = v.x + v.y + v.z + v.w; __syncthreads();
    for (int o = 128; o > 0; o >>= 1) { if (tid < o) red[tid] += red[tid + o]; __syncthreads(); }
    const float mean = red[0] * (1.0f / 1024.0f);
    __syncthreads();

    float qx = v.x - mean, qy = v.y - mean, qz = v.z - mean, qw = v.w - mean;
    red[tid] = qx * qx + qy * qy + qz * qz + qw * qw; __syncthreads();
    for (int o = 128; o > 0; o >>= 1) { if (tid < o) red[tid] += red[tid + o]; __syncthreads(); }
    const float rstd = rsqrtf(red[0] * (1.0f / 1024.0f) + 1e-6f);

    const int bt = r >> 8;
    const float4 sh = *(const float4*)(modbuf + (size_t)bt * mod_ld + shift_off + tid * 4);
    const float4 sc = *(const float4*)(modbuf + (size_t)bt * mod_ld + scale_off + tid * 4);

    int orow = r;
    if (transposed) {
        int b = r >> 12, t = (r >> 8) & 15, sp = r & 255;
        orow = ((b << 8) + sp) * 16 + t;
    }
    v4bf o4 = { (bf16_t)(qx * rstd * (1.0f + sc.x) + sh.x),
                (bf16_t)(qy * rstd * (1.0f + sc.y) + sh.y),
                (bf16_t)(qz * rstd * (1.0f + sc.z) + sh.z),
                (bf16_t)(qw * rstd * (1.0f + sc.w) + sh.w) };
    *(v4bf*)(out + (size_t)orow * 1024 + tid * 4) = o4;
}

// out[r,d] = x[r,d] + a[src(r),d] * gate[bt(r),d]   (4 lanes per thread, f32)
__global__ void add_gate_kernel(
    const float* __restrict__ x, const float* __restrict__ a,
    const float* __restrict__ modbuf, int mod_ld, int gate_off,
    float* __restrict__ out, int transposed, int n4)
{
    int idx = blockIdx.x * blockDim.x + threadIdx.x;
    if (idx >= n4) return;
    int r = idx >> 8;               // 256 float4 per 1024-row
    int d4 = (idx & 255) << 2;
    int bt = r >> 8;
    int src = r;
    if (transposed) {
        int b = r >> 12, t = (r >> 8) & 15, sp = r & 255;
        src = ((b << 8) + sp) * 16 + t;
    }
    float4 xv = *(const float4*)(x + (size_t)r * 1024 + d4);
    float4 av = *(const float4*)(a + (size_t)src * 1024 + d4);
    float4 gv = *(const float4*)(modbuf + (size_t)bt * mod_ld + gate_off + d4);
    float4 o4 = make_float4(xv.x + av.x * gv.x, xv.y + av.y * gv.y,
                            xv.z + av.z * gv.z, xv.w + av.w * gv.w);
    *(float4*)(out + (size_t)r * 1024 + d4) = o4;
}

// row-wise softmax: f32 scores in, bf16 probabilities out; one block per row
__global__ __launch_bounds__(256) void softmax_kernel(
    const float* __restrict__ S, bf16_t* __restrict__ P, int L)
{
    const float* p = S + (size_t)blockIdx.x * L;
    bf16_t* q = P + (size_t)blockIdx.x * L;
    __shared__ float red[256];
    const int tid = threadIdx.x;

    float m = -3.402823466e38f;
    for (int i = tid; i < L; i += 256) m = fmaxf(m, p[i]);
    red[tid] = m; __syncthreads();
    for (int o = 128; o > 0; o >>= 1) { if (tid < o) red[tid] = fmaxf(red[tid], red[tid + o]); __syncthreads(); }
    m = red[0]; __syncthreads();

    float s = 0.0f;
    for (int i = tid; i < L; i += 256) s += __expf(p[i] - m);
    red[tid] = s; __syncthreads();
    for (int o = 128; o > 0; o >>= 1) { if (tid < o) red[tid] += red[tid + o]; __syncthreads(); }
    const float inv = 1.0f / red[0];
    for (int i = tid; i < L; i += 256) q[i] = (bf16_t)(__expf(p[i] - m) * inv);
}

// LSTM cell pointwise (4 hidden units/thread): gates = xg[n*16+t] + gtmp[n].
// Writes f32 cell state, bf16 h (next GEMM input), f32 h into the output seq.
__global__ void lstm_cell_kernel(
    const float* __restrict__ xg, const float* __restrict__ gtmp,
    bf16_t* __restrict__ hb, float* __restrict__ c,
    float* __restrict__ out, int t)
{
    int idx = blockIdx.x * blockDim.x + threadIdx.x;   // 512*256 threads
    if (idx >= 512 * 256) return;
    int n = idx >> 8, j = (idx & 255) << 2;
    const float* xr = xg + (size_t)(n * 16 + t) * 4096;
    const float* gr = gtmp + (size_t)n * 4096;
    float4 gi = *(const float4*)(xr + j);
    float4 gf = *(const float4*)(xr + 1024 + j);
    float4 gg = *(const float4*)(xr + 2048 + j);
    float4 go = *(const float4*)(xr + 3072 + j);
    float4 ri = *(const float4*)(gr + j);
    float4 rf = *(const float4*)(gr + 1024 + j);
    float4 rg = *(const float4*)(gr + 2048 + j);
    float4 ro = *(const float4*)(gr + 3072 + j);
    float4 cv = *(const float4*)(c + (size_t)n * 1024 + j);
    float4 cc, hh;
    cc.x = sigm(gf.x + rf.x) * cv.x + sigm(gi.x + ri.x) * tanhf(gg.x + rg.x);
    cc.y = sigm(gf.y + rf.y) * cv.y + sigm(gi.y + ri.y) * tanhf(gg.y + rg.y);
    cc.z = sigm(gf.z + rf.z) * cv.z + sigm(gi.z + ri.z) * tanhf(gg.z + rg.z);
    cc.w = sigm(gf.w + rf.w) * cv.w + sigm(gi.w + ri.w) * tanhf(gg.w + rg.w);
    hh.x = sigm(go.x + ro.x) * tanhf(cc.x);
    hh.y = sigm(go.y + ro.y) * tanhf(cc.y);
    hh.z = sigm(go.z + ro.z) * tanhf(cc.z);
    hh.w = sigm(go.w + ro.w) * tanhf(cc.w);
    *(float4*)(c + (size_t)n * 1024 + j) = cc;
    v4bf hb4 = { (bf16_t)hh.x, (bf16_t)hh.y, (bf16_t)hh.z, (bf16_t)hh.w };
    *(v4bf*)(hb + (size_t)n * 1024 + j) = hb4;
    *(float4*)(out + (size_t)(n * 16 + t) * 1024 + j) = hh;
}

// ---------------------------------------------------------------------------
// Host-side launchers
// ---------------------------------------------------------------------------
static inline void gemm(hipStream_t s, const bf16_t* A, int lda, const bf16_t* BT,
                        int ldb, const float* b0, const float* b1,
                        float* Cf, bf16_t* Cb, int ldc, int M, int N, int K, int act)
{
    dim3 g((N + 127) / 128, (M + 127) / 128);
    if (Cb) {
        if (act)
            gemm_kernel<1, 1><<<g, dim3(256), 0, s>>>(A, lda, BT, ldb, b0, b1, nullptr, Cb, ldc, M, N, K);
        else
            gemm_kernel<0, 1><<<g, dim3(256), 0, s>>>(A, lda, BT, ldb, b0, b1, nullptr, Cb, ldc, M, N, K);
    } else {
        gemm_kernel<0, 0><<<g, dim3(256), 0, s>>>(A, lda, BT, ldb, b0, b1, Cf, nullptr, ldc, M, N, K);
    }
}

static inline void wconv(hipStream_t s, const float* W, bf16_t* WT, int K, int N)
{
    wconv_kernel<<<dim3(N / 64, K / 64), dim3(256), 0, s>>>(W, WT, K, N);
}

extern "C" void kernel_launch(void* const* d_in, const int* in_sizes, int n_in,
                              void* d_out, int out_size, void* d_ws, size_t ws_size,
                              hipStream_t stream)
{
    (void)in_sizes; (void)n_in; (void)out_size; (void)ws_size;

    const float* x        = (const float*)d_in[0];
    const float* c        = (const float*)d_in[1];
    const float* s_ada_w  = (const float*)d_in[2];
    const float* s_ada_b  = (const float*)d_in[3];
    const float* t_ada_w  = (const float*)d_in[4];
    const float* t_ada_b  = (const float*)d_in[5];
    const float* r_ada_w  = (const float*)d_in[6];
    const float* r_ada_b  = (const float*)d_in[7];
    const float* s_qkv_w  = (const float*)d_in[8];
    const float* s_out_w  = (const float*)d_in[9];
    const float* s_out_b  = (const float*)d_in[10];
    const float* t_qkv_w  = (const float*)d_in[11];
    const float* t_out_w  = (const float*)d_in[12];
    const float* t_out_b  = (const float*)d_in[13];
    const float* s_mlp_w1 = (const float*)d_in[14];
    const float* s_mlp_b1 = (const float*)d_in[15];
    const float* s_mlp_w2 = (const float*)d_in[16];
    const float* s_mlp_b2 = (const float*)d_in[17];
    const float* t_mlp_w1 = (const float*)d_in[18];
    const float* t_mlp_b1 = (const float*)d_in[19];
    const float* t_mlp_w2 = (const float*)d_in[20];
    const float* t_mlp_b2 = (const float*)d_in[21];
    const float* w_ih     = (const float*)d_in[22];
    const float* w_hh     = (const float*)d_in[23];
    const float* b_ih     = (const float*)d_in[24];
    const float* b_hh     = (const float*)d_in[25];
    float* out = (float*)d_out;

    // ---- workspace carve-out: f32 region first, then bf16 region ----
    float* wf = (float*)d_ws;
    auto takef = [&](size_t n) { float* p = wf; wf += n; return p; };
    float* smod  = takef((size_t)32 * 6144);
    float* tmod  = takef((size_t)32 * 6144);
    float* rmod  = takef((size_t)32 * 3072);
    float* xcur  = takef((size_t)8192 * 1024);
    float* bproj = takef((size_t)8192 * 1024);
    float* bbig  = takef((size_t)33554432);   // f32 scores / LSTM pre-gates
    float* gtmp  = takef((size_t)512 * 4096);
    float* cst   = takef((size_t)512 * 1024);

    bf16_t* wb = (bf16_t*)wf;
    auto takeb = [&](size_t n) { bf16_t* p = wb; wb += n; return p; };
    bf16_t* csb    = takeb((size_t)32 * 1024);
    bf16_t* blnb   = takeb((size_t)8192 * 1024);
    bf16_t* bqkvb  = takeb((size_t)8192 * 3072);
    bf16_t* battnb = takeb((size_t)8192 * 1024);
    bf16_t* bbigb  = takeb((size_t)33554432);  // bf16 P / MLP hidden
    bf16_t* hb     = takeb((size_t)512 * 1024);
    bf16_t* sadawt = takeb((size_t)1024 * 6144);
    bf16_t* tadawt = takeb((size_t)1024 * 6144);
    bf16_t* radawt = takeb((size_t)1024 * 3072);
    bf16_t* sqkvwt = takeb((size_t)1024 * 3072);
    bf16_t* soutwt = takeb((size_t)1024 * 1024);
    bf16_t* tqkvwt = takeb((size_t)1024 * 3072);
    bf16_t* toutwt = takeb((size_t)1024 * 1024);
    bf16_t* smlp1t = takeb((size_t)1024 * 4096);
    bf16_t* smlp2t = takeb((size_t)4096 * 1024);
    bf16_t* tmlp1t = takeb((size_t)1024 * 4096);
    bf16_t* tmlp2t = takeb((size_t)4096 * 1024);
    bf16_t* wiht   = takeb((size_t)1024 * 4096);
    bf16_t* whht   = takeb((size_t)1024 * 4096);

    const int ELT4 = 8192 * 1024 / 4;
    const int EB4  = (ELT4 + 255) / 256;

    // ---- one-time per-call: weights -> bf16 [N,K] ----
    wconv(stream, s_ada_w, sadawt, 1024, 6144);
    wconv(stream, t_ada_w, tadawt, 1024, 6144);
    wconv(stream, r_ada_w, radawt, 1024, 3072);
    wconv(stream, s_qkv_w, sqkvwt, 1024, 3072);
    wconv(stream, s_out_w, soutwt, 1024, 1024);
    wconv(stream, t_qkv_w, tqkvwt, 1024, 3072);
    wconv(stream, t_out_w, toutwt, 1024, 1024);
    wconv(stream, s_mlp_w1, smlp1t, 1024, 4096);
    wconv(stream, s_mlp_w2, smlp2t, 4096, 1024);
    wconv(stream, t_mlp_w1, tmlp1t, 1024, 4096);
    wconv(stream, t_mlp_w2, tmlp2t, 4096, 1024);
    wconv(stream, w_ih, wiht, 1024, 4096);
    wconv(stream, w_hh, whht, 1024, 4096);

    // conditioning
    silu_kernel<<<(8192 + 255) / 256, 256, 0, stream>>>((const float4*)c, csb, 8192);
    gemm(stream, csb, 1024, sadawt, 1024, s_ada_b, nullptr, smod, nullptr, 6144, 32, 6144, 1024, 0);
    gemm(stream, csb, 1024, tadawt, 1024, t_ada_b, nullptr, tmod, nullptr, 6144, 32, 6144, 1024, 0);
    gemm(stream, csb, 1024, radawt, 1024, r_ada_b, nullptr, rmod, nullptr, 3072, 32, 3072, 1024, 0);
    copy_kernel<<<EB4, 256, 0, stream>>>((const float4*)x, (float4*)xcur, ELT4);

    // ---- spatial attention (L=256, 32 instances, non-causal) ----
    ln_mod_kernel<<<8192, 256, 0, stream>>>(xcur, smod, 6144, 0, 1024, blnb, 0);
    gemm(stream, blnb, 1024, sqkvwt, 1024, nullptr, nullptr, nullptr, bqkvb, 3072, 8192, 3072, 1024, 0);
    attn_score_kernel<false><<<dim3(4, 4, 512), 128, 0, stream>>>(bqkvb, 256, bbig);
    softmax_kernel<<<512 * 256, 256, 0, stream>>>(bbig, bbigb, 256);
    attn_av_kernel<<<dim3(1, 4, 512), 128, 0, stream>>>(bbigb, bqkvb, 256, battnb);
    gemm(stream, battnb, 1024, soutwt, 1024, s_out_b, nullptr, bproj, nullptr, 1024, 8192, 1024, 1024, 0);
    add_gate_kernel<<<EB4, 256, 0, stream>>>(xcur, bproj, smod, 6144, 2048, xcur, 0, ELT4);

    // ---- spatial MLP ----
    ln_mod_kernel<<<8192, 256, 0, stream>>>(xcur, smod, 6144, 3072, 4096, blnb, 0);
    gemm(stream, blnb, 1024, smlp1t, 1024, s_mlp_b1, nullptr, nullptr, bbigb, 4096, 8192, 4096, 1024, 1);
    gemm(stream, bbigb, 4096, smlp2t, 4096, s_mlp_b2, nullptr, bproj, nullptr, 1024, 8192, 1024, 4096, 0);
    add_gate_kernel<<<EB4, 256, 0, stream>>>(xcur, bproj, smod, 6144, 5120, xcur, 0, ELT4);

    // ---- temporal attention (L=16, 512 instances, causal; transposed order) ----
    ln_mod_kernel<<<8192, 256, 0, stream>>>(xcur, tmod, 6144, 0, 1024, blnb, 1);
    gemm(stream, blnb, 1024, tqkvwt, 1024, nullptr, nullptr, nullptr, bqkvb, 3072, 8192, 3072, 1024, 0);
    attn_score_kernel<true><<<dim3(1, 1, 8192), 128, 0, stream>>>(bqkvb, 16, bbig);
    softmax_kernel<<<8192 * 16, 256, 0, stream>>>(bbig, bbigb, 16);
    attn_av_kernel<<<dim3(1, 1, 8192), 128, 0, stream>>>(bbigb, bqkvb, 16, battnb);
    gemm(stream, battnb, 1024, toutwt, 1024, t_out_b, nullptr, bproj, nullptr, 1024, 8192, 1024, 1024, 0);
    add_gate_kernel<<<EB4, 256, 0, stream>>>(xcur, bproj, tmod, 6144, 2048, xcur, 1, ELT4);

    // ---- temporal MLP ----
    ln_mod_kernel<<<8192, 256, 0, stream>>>(xcur, tmod, 6144, 3072, 4096, blnb, 0);
    gemm(stream, blnb, 1024, tmlp1t, 1024, t_mlp_b1, nullptr, nullptr, bbigb, 4096, 8192, 4096, 1024, 1);
    gemm(stream, bbigb, 4096, tmlp2t, 4096, t_mlp_b2, nullptr, bproj, nullptr, 1024, 8192, 1024, 4096, 0);
    add_gate_kernel<<<EB4, 256, 0, stream>>>(xcur, bproj, tmod, 6144, 5120, xcur, 0, ELT4);

    // ---- recurrent branch (LSTM over T per spatial token) ----
    ln_mod_kernel<<<8192, 256, 0, stream>>>(xcur, rmod, 3072, 0, 1024, blnb, 1);
    // xg = xr @ w_ih + (b_ih + b_hh): one big WMMA GEMM, f32 out
    gemm(stream, blnb, 1024, wiht, 1024, b_ih, b_hh, bbig, nullptr, 4096, 8192, 4096, 1024, 0);
    zero_kernel<<<(65536 + 255) / 256, 256, 0, stream>>>((float4*)hb, 65536);       // 512*1024 bf16
    zero_kernel<<<(512 * 256 + 255) / 256, 256, 0, stream>>>((float4*)cst, 512 * 256);
    for (int t = 0; t < 16; ++t) {
        gemm(stream, hb, 1024, whht, 1024, nullptr, nullptr, gtmp, nullptr, 4096, 512, 4096, 1024, 0);
        lstm_cell_kernel<<<(512 * 256 + 255) / 256, 256, 0, stream>>>(bbig, gtmp, hb, cst, bproj, t);
    }
    add_gate_kernel<<<EB4, 256, 0, stream>>>(xcur, bproj, rmod, 3072, 2048, out, 1, ELT4);
}
